// GF1_23562190586432
// MI455X (gfx1250) — matrix operand order repaired and checked by reading, not compiled, not run on hardware
//
#include <hip/hip_runtime.h>
#include <hip/hip_bf16.h>
#include <stdint.h>

// ---------------------------------------------------------------------------
// Fixed problem sizes from the reference
// ---------------------------------------------------------------------------
#define BB   4
#define DIM  64
#define HH   64
#define WW   64
#define LL   (HH*WW)          // 4096
#define PP   (BB*LL)          // 16384 positions
#define HID  170
#define DIN  340
#define DST  16
#define DTR  11
#define KK   4

typedef __attribute__((ext_vector_type(16))) _Float16 v16h;
typedef __attribute__((ext_vector_type(8)))  float    v8f;

union FragU { uint4 u[2]; v16h h; };

// ---------------------------------------------------------------------------
// Generic NT WMMA GEMM:  C[M,N] = A[M,Kp] * B[N,Kp]^T  (+ bias[col]) (+ add)
// A,B f16 with K contiguous; Kp multiple of 32; M multiple of 16.
// B must be padded to ceil64(N) rows (rows >= N may hold garbage; they only
// feed masked output columns).  One wave -> one 16x64 C strip (4 tiles):
// the A fragment is loaded once per K-step and reused for 4 WMMAs, cutting
// per-WMMA load traffic from 4 to 2.5 b128 loads.
// Fragment k order: contiguous, lane half h holds k = 16h..16h+15 in A and B alike (f32 C/D per ISA).
// ---------------------------------------------------------------------------
__global__ void wmma_nt_gemm(const _Float16* __restrict__ A,
                             const _Float16* __restrict__ Bm,
                             float* __restrict__ C,
                             const float* __restrict__ bias,
                             const float* __restrict__ add,
                             int M, int N, int Kp,
                             int lda, int ldb, int ldc,
                             long long sA, long long sB, long long sC,
                             long long sAdd)
{
    const int bt = blockIdx.z;
    const _Float16* Ab = A  + (long long)bt * sA;
    const _Float16* Bb = Bm + (long long)bt * sB;
    float*          Cb = C  + (long long)bt * sC;
    const float*    Db = add ? add + (long long)bt * sAdd : nullptr;

    const int lane  = threadIdx.x;        // wave32
    const int m0    = blockIdx.y << 4;
    const int n0    = blockIdx.x << 6;    // 64 columns per wave
    const int arow  = m0 + (lane & 15);
    const int koffA = (lane >> 4) << 4;   // 0 or 16
    const int bcol0 = n0 + (lane & 15);
    const int koffB = (lane >> 4) << 4;   // 0 or 16

    v8f acc0 = {}, acc1 = {}, acc2 = {}, acc3 = {};
    const _Float16* aRow = Ab + (long long)arow * lda + koffA;
    const _Float16* bRow = Bb + (long long)bcol0 * ldb + koffB;
    const long long bTile = (long long)16 * ldb;

    for (int kk = 0; kk < Kp; kk += 32) {
        // prefetch 8 K-steps (256 halves) ahead (speculative, gfx1250 global_prefetch_b8)
        __builtin_prefetch((const void*)(aRow + kk + 256), 0, 0);
        __builtin_prefetch((const void*)(bRow + kk + 256), 0, 0);

        FragU a, b0, b1, b2, b3;
        const _Float16* ap = aRow + kk;
        a.u[0] = *(const uint4*)(ap);
        a.u[1] = *(const uint4*)(ap + 8);
        const _Float16* bp = bRow + kk;
        b0.u[0] = ((const uint4*)(bp            ))[0];
        b0.u[1] = ((const uint4*)(bp            ))[1];
        b1.u[0] = ((const uint4*)(bp + bTile    ))[0];
        b1.u[1] = ((const uint4*)(bp + bTile    ))[1];
        b2.u[0] = ((const uint4*)(bp + 2 * bTile))[0];
        b2.u[1] = ((const uint4*)(bp + 2 * bTile))[1];
        b3.u[0] = ((const uint4*)(bp + 3 * bTile))[0];
        b3.u[1] = ((const uint4*)(bp + 3 * bTile))[1];

        acc0 = __builtin_amdgcn_wmma_f32_16x16x32_f16(false, a.h, false, b0.h, (short)0, acc0, false, false);
        acc1 = __builtin_amdgcn_wmma_f32_16x16x32_f16(false, a.h, false, b1.h, (short)0, acc1, false, false);
        acc2 = __builtin_amdgcn_wmma_f32_16x16x32_f16(false, a.h, false, b2.h, (short)0, acc2, false, false);
        acc3 = __builtin_amdgcn_wmma_f32_16x16x32_f16(false, a.h, false, b3.h, (short)0, acc3, false, false);
    }

    const int rbase = m0 + ((lane >> 4) << 3);
#pragma unroll
    for (int t = 0; t < 4; ++t) {
        const int col = n0 + (t << 4) + (lane & 15);
        if (col >= N) continue;
        const float bv = bias ? bias[col] : 0.0f;
        const v8f acc = (t == 0) ? acc0 : (t == 1) ? acc1 : (t == 2) ? acc2 : acc3;
#pragma unroll
        for (int v = 0; v < 8; ++v) {
            const int row = rbase + v;
            if (row < M) {
                const long long idx = (long long)row * ldc + col;
                float r = acc[v] + bv;
                if (Db) r += Db[idx];
                Cb[idx] = r;
            }
        }
    }
}

// ---------------------------------------------------------------------------
// Layout / packing helpers
// ---------------------------------------------------------------------------

// NCHW f32 -> NHWC f32  (input_R -> Rcl)
__global__ void k_nchw_to_nhwc(const float* __restrict__ src, float* __restrict__ dst)
{
    long long i = (long long)blockIdx.x * blockDim.x + threadIdx.x;
    if (i >= (long long)PP * DIM) return;
    int c = (int)(i & 63); long long t = i >> 6;
    int p = (int)(t & (LL - 1)); int b = (int)(t >> 12);
    dst[i] = src[((long long)(b * DIM + c)) * LL + p];
}

// NHWC f32 -> NCHW f32 (final output)
__global__ void k_nhwc_to_nchw(const float* __restrict__ src, float* __restrict__ dst)
{
    long long i = (long long)blockIdx.x * blockDim.x + threadIdx.x;
    if (i >= (long long)PP * DIM) return;
    int p = (int)(i & (LL - 1)); long long t = i >> 12;
    int c = (int)(t & 63); int b = (int)(t >> 6);
    dst[i] = src[((long long)(b * LL + p)) * DIM + c];
}

// generic f32 (R,Cc) strided -> f16 (Rp,Kp) zero-padded
__global__ void k_pack16(const float* __restrict__ src, int R, int Cc, int lds,
                         _Float16* __restrict__ dst, int Rp, int Kp)
{
    long long i = (long long)blockIdx.x * blockDim.x + threadIdx.x;
    long long total = (long long)Rp * Kp;
    if (i >= total) return;
    int c = (int)(i % Kp); long long r = i / Kp;
    dst[i] = (r < R && c < Cc) ? (_Float16)src[r * lds + c] : (_Float16)0.0f;
}

// nearest upsample of input_S (B,32,32,32) -> f16 (P,32) channels-last
__global__ void k_up_pack(const float* __restrict__ S, _Float16* __restrict__ dst)
{
    long long i = (long long)blockIdx.x * blockDim.x + threadIdx.x;
    if (i >= (long long)PP * 32) return;
    int ic = (int)(i & 31); long long t = i >> 5;
    int x = (int)(t & 63); int y = (int)((t >> 6) & 63); int b = (int)(t >> 12);
    dst[i] = (_Float16)S[(((long long)(b * 32 + ic)) * 32 + (y >> 1)) * 32 + (x >> 1)];
}

// LayerNorm over last (contiguous) C channels -> f16 padded to Kp
__global__ void k_ln_pack(const float* __restrict__ src, int ld, int C,
                          const float* __restrict__ w, const float* __restrict__ bvec,
                          float eps, _Float16* __restrict__ dst, int Kp, int rows)
{
    int r = blockIdx.x * blockDim.x + threadIdx.x;
    if (r >= rows) return;
    const float* s = src + (long long)r * ld;
    float mu = 0.f;
    for (int c = 0; c < C; ++c) mu += s[c];
    mu /= (float)C;
    float var = 0.f;
    for (int c = 0; c < C; ++c) { float d = s[c] - mu; var += d * d; }
    var /= (float)C;
    float inv = rsqrtf(var + eps);
    _Float16* o = dst + (long long)r * Kp;
    for (int c = 0; c < C; ++c) o[c] = (_Float16)((s[c] - mu) * inv * w[c] + bvec[c]);
    for (int c = C; c < Kp; ++c) o[c] = (_Float16)0.0f;
}

// depthwise 3x3 SAME, channels-last.  act: 0 none, 1 silu(+bias), 2 tanh+res
__global__ void k_dw3x3(const float* __restrict__ src, int lds,
                        const float* __restrict__ wdw,
                        const float* __restrict__ bias,
                        const float* __restrict__ res, int ldr,
                        float* __restrict__ dst, int ldd, int C, int act)
{
    long long i = (long long)blockIdx.x * blockDim.x + threadIdx.x;
    long long total = (long long)PP * C;
    if (i >= total) return;
    int c = (int)(i % C); long long t = i / C;
    int x = (int)(t & 63); int y = (int)((t >> 6) & 63); int b = (int)(t >> 12);
    float acc = 0.f;
#pragma unroll
    for (int ky = 0; ky < 3; ++ky) {
        int yy = y + ky - 1;
        if (yy < 0 || yy >= HH) continue;
#pragma unroll
        for (int kx = 0; kx < 3; ++kx) {
            int xx = x + kx - 1;
            if (xx < 0 || xx >= WW) continue;
            acc += src[((long long)(b * LL + yy * WW + xx)) * lds + c] * wdw[c * 9 + ky * 3 + kx];
        }
    }
    if (bias) acc += bias[c];
    if (act == 1) acc = acc / (1.0f + __expf(-acc));                       // silu
    else if (act == 2) acc = tanhf(acc) + res[((long long)t) * ldr + c];   // tanh + residual
    dst[((long long)t) * ldd + c] = acc;
}

// per-(b,channel) L2 norm over L, write transposed f16 row-major (bc, l)
__global__ void k_l2n_pack(const float* __restrict__ src, int ld, int coff,
                           _Float16* __restrict__ dst)
{
    const int bc = blockIdx.x;            // b*64 + c, 256 rows
    const int b = bc >> 6, c = bc & 63;
    const int tid = threadIdx.x;
    __shared__ float red[256];
    float ss = 0.f;
    for (int l = tid; l < LL; l += 256) {
        float v = src[((long long)(b * LL + l)) * ld + coff + c];
        ss += v * v;
    }
    red[tid] = ss;
    __syncthreads();
    for (int s = 128; s > 0; s >>= 1) {
        if (tid < s) red[tid] += red[tid + s];
        __syncthreads();
    }
    float inv = 1.0f / fmaxf(sqrtf(red[0]), 1e-12f);
    for (int l = tid; l < LL; l += 256) {
        float v = src[((long long)(b * LL + l)) * ld + coff + c];
        dst[(long long)bc * LL + l] = (_Float16)(v * inv);
    }
}

// v rearrange: kv1 (P,128) ch 64..127 -> f16 (b,h,l,32)
__global__ void k_pack_v(const float* __restrict__ kv1, _Float16* __restrict__ dst)
{
    long long i = (long long)blockIdx.x * blockDim.x + threadIdx.x;
    if (i >= (long long)BB * 2 * LL * 32) return;
    int d = (int)(i & 31); long long t = i >> 5;
    int l = (int)(t & (LL - 1)); int bh = (int)(t >> 12);
    int b = bh >> 1, h = bh & 1;
    dst[i] = (_Float16)kv1[((long long)(b * LL + l)) * 128 + 64 + h * 32 + d];
}

// softmax over 32 with temperature; one wave per row
// out layout: 64-row padded batches -> out[bh*2048 + c*32 + d]
__global__ void k_softmax32(const float* __restrict__ attn,
                            const float* __restrict__ temp,
                            _Float16* __restrict__ out)
{
    int row = blockIdx.x * 8 + (threadIdx.x >> 5);
    int d = threadIdx.x & 31;
    if (row >= BB * 2 * 32) return;
    int h = (row >> 5) & 1;
    int bh = row >> 5, c = row & 31;
    float v = attn[row * 32 + d] * temp[h];
    float mx = v;
    for (int o = 16; o > 0; o >>= 1) mx = fmaxf(mx, __shfl_xor(mx, o, 32));
    float e = __expf(v - mx);
    float sm = e;
    for (int o = 16; o > 0; o >>= 1) sm += __shfl_xor(sm, o, 32);
    out[bh * 2048 + c * 32 + d] = (_Float16)(e / sm);
}

// (b,h,l,32) f32 -> f16 (P,64) channels-last, heads merged
__global__ void k_repack_heads(const float* __restrict__ g2, _Float16* __restrict__ dst)
{
    long long i = (long long)blockIdx.x * blockDim.x + threadIdx.x;
    if (i >= (long long)PP * DIM) return;
    int c64 = (int)(i & 63); long long t = i >> 6;
    int l = (int)(t & (LL - 1)); int b = (int)(t >> 12);
    int h = c64 >> 5, c = c64 & 31;
    dst[i] = (_Float16)g2[(((long long)(b * 2 + h) * LL + l) << 5) + c];
}

// scan-direction index map: scan index l (dir k) -> spatial index m
__device__ __forceinline__ int dirmap(int k, int l)
{
    if (k == 0) return l;
    if (k == 1) return ((l & 63) << 6) | (l >> 6);
    if (k == 2) return (LL - 1) - l;
    int lr = (LL - 1) - l;
    return ((lr & 63) << 6) | (lr >> 6);
}

// build xs f16 (b,k,l,352) from xc (P,340) with direction gather
__global__ void k_pack_xs(const float* __restrict__ xc, _Float16* __restrict__ dst)
{
    long long i = (long long)blockIdx.x * blockDim.x + threadIdx.x;
    long long total = (long long)BB * KK * LL * 352;
    if (i >= total) return;
    int d = (int)(i % 352); long long t = i / 352;
    int l = (int)(t & (LL - 1)); int bk = (int)(t >> 12);
    int b = bk >> 2, k = bk & 3;
    if (d >= DIN) { dst[i] = (_Float16)0.0f; return; }
    int m = dirmap(k, l);
    dst[i] = (_Float16)xc[((long long)(b * LL + m)) * DIN + d];
}

// replicate ss_xproj_w (K,43,340) -> f16 (b*4+k, 64, 352) zero-padded
__global__ void k_pack_xpw(const float* __restrict__ xpw, _Float16* __restrict__ dst)
{
    long long i = (long long)blockIdx.x * blockDim.x + threadIdx.x;
    long long total = (long long)BB * KK * 64 * 352;
    if (i >= total) return;
    int d = (int)(i % 352); long long t = i / 352;
    int c = (int)(t % 64); int bk = (int)(t / 64); int k = bk & 3;
    dst[i] = (c < (DTR + 2 * DST) && d < DIN)
               ? (_Float16)xpw[((long long)(k * (DTR + 2 * DST) + c)) * DIN + d]
               : (_Float16)0.0f;
}

// ---------------------------------------------------------------------------
// Selective scan: one thread per (b,k,d), 16 states in registers, L=4096 steps.
// Fuses dt projection (11 MACs) + softplus + D*u skip term.
// ---------------------------------------------------------------------------
__global__ void k_selscan(const float* __restrict__ xc,
                          const float* __restrict__ xdbl,   // (b,k,l,43)
                          const float* __restrict__ dtw,    // (K,340,11)
                          const float* __restrict__ dtb,    // (K,340)
                          const float* __restrict__ Alogs,  // (K*340,16)
                          const float* __restrict__ Ds,     // (K*340)
                          float* __restrict__ oy)           // (b,k,l,340)
{
    const int bk = blockIdx.x;            // 0..15
    const int b = bk >> 2, k = bk & 3;
    const int d = threadIdx.x;
    if (d >= DIN) return;

    float Areg[DST], h[DST];
#pragma unroll
    for (int n = 0; n < DST; ++n) {
        Areg[n] = -__expf(Alogs[(((long long)(k * DIN + d)) << 4) + n]);
        h[n] = 0.0f;
    }
    float wdt[DTR];
#pragma unroll
    for (int r = 0; r < DTR; ++r) wdt[r] = dtw[((long long)(k * DIN + d)) * DTR + r];
    const float bdt = dtb[k * DIN + d];
    const float Dv  = Ds[k * DIN + d];

    const float* xdb = xdbl + (long long)bk * LL * 43;
    float*       oyb = oy   + (long long)bk * LL * DIN;

    for (int l = 0; l < LL; ++l) {
        const int m = dirmap(k, l);
        const float u = xc[((long long)(b * LL + m)) * DIN + d];
        const float* row = xdb + (long long)l * 43;
        float dl = bdt;
#pragma unroll
        for (int r = 0; r < DTR; ++r) dl += row[r] * wdt[r];
        dl = (dl > 20.0f) ? dl : log1pf(__expf(dl));   // softplus
        const float du = dl * u;
        float y = 0.0f;
#pragma unroll
        for (int n = 0; n < DST; ++n) {
            float hn = __expf(dl * Areg[n]) * h[n] + du * row[DTR + n];
            h[n] = hn;
            y += hn * row[DTR + DST + n];
        }
        oyb[(long long)l * DIN + d] = y + Dv * u;
    }
}

// combine the 4 scan directions back to spatial order: ycomb (P,340)
__global__ void k_combine(const float* __restrict__ oy, float* __restrict__ yc)
{
    long long i = (long long)blockIdx.x * blockDim.x + threadIdx.x;
    if (i >= (long long)PP * DIN) return;
    int d = (int)(i % DIN); long long t = i / DIN;
    int m = (int)(t & (LL - 1)); int b = (int)(t >> 12);
    int l1 = ((m & 63) << 6) | (m >> 6);
    long long base = (long long)b * KK * LL * DIN;
    float v = oy[base + ((long long)(0 * LL) + m) * DIN + d]
            + oy[base + ((long long)(1 * LL) + l1) * DIN + d]
            + oy[base + ((long long)(2 * LL) + (LL - 1 - m)) * DIN + d]
            + oy[base + ((long long)(3 * LL) + (LL - 1 - l1)) * DIN + d];
    yc[i] = v;
}

// ln over 340 (onorm, eps 1e-5) * silu(z) -> f16 (P,352) padded
__global__ void k_ln_silu_pack(const float* __restrict__ y,
                               const float* __restrict__ xxz,
                               const float* __restrict__ w,
                               const float* __restrict__ bvec,
                               _Float16* __restrict__ dst)
{
    int r = blockIdx.x * blockDim.x + threadIdx.x;
    if (r >= PP) return;
    const float* s = y + (long long)r * DIN;
    float mu = 0.f;
    for (int c = 0; c < DIN; ++c) mu += s[c];
    mu /= (float)DIN;
    float var = 0.f;
    for (int c = 0; c < DIN; ++c) { float dd = s[c] - mu; var += dd * dd; }
    var /= (float)DIN;
    float inv = rsqrtf(var + 1e-5f);
    _Float16* o = dst + (long long)r * 352;
    const float* z = xxz + (long long)r * (2 * DIN) + DIN;
    for (int c = 0; c < DIN; ++c) {
        float ln = (s[c] - mu) * inv * w[c] + bvec[c];
        float zz = z[c];
        o[c] = (_Float16)(ln * (zz / (1.0f + __expf(-zz))));
    }
    for (int c = DIN; c < 352; ++c) o[c] = (_Float16)0.0f;
}

// x1f = tanh(ssout)+x1 ; prod = x1f * x2b -> f16 (P,192) padded
__global__ void k_eltfinal(const float* __restrict__ ssout,
                           const float* __restrict__ dwout,
                           const float* __restrict__ x2b,
                           _Float16* __restrict__ dst)
{
    long long i = (long long)blockIdx.x * blockDim.x + threadIdx.x;
    if (i >= (long long)PP * 192) return;
    int c = (int)(i % 192); long long r = i / 192;
    if (c >= HID) { dst[i] = (_Float16)0.0f; return; }
    float x1f = tanhf(ssout[r * HID + c]) + dwout[r * DIN + c];
    dst[i] = (_Float16)(x1f * x2b[r * HID + c]);
}

// ---------------------------------------------------------------------------
// Host orchestration
// ---------------------------------------------------------------------------
static inline long long cdiv(long long a, long long b) { return (a + b - 1) / b; }

static void gemm(hipStream_t st, const _Float16* A, const _Float16* B, float* C,
                 const float* bias, const float* add,
                 int M, int N, int Kp, int lda, int ldb, int ldc,
                 long long sA, long long sB, long long sC, long long sAdd, int batch)
{
    dim3 grid((unsigned)cdiv(N, 64), (unsigned)(M / 16), (unsigned)batch);
    wmma_nt_gemm<<<grid, dim3(32), 0, st>>>(A, B, C, bias, add, M, N, Kp,
                                            lda, ldb, ldc, sA, sB, sC, sAdd);
}

extern "C" void kernel_launch(void* const* d_in, const int* in_sizes, int n_in,
                              void* d_out, int out_size, void* d_ws, size_t ws_size,
                              hipStream_t stream)
{
    const float* input_R   = (const float*)d_in[0];
    const float* input_S   = (const float*)d_in[1];
    const float* conv1_w   = (const float*)d_in[2];
    const float* conv1_b   = (const float*)d_in[3];
    const float* norm1_w   = (const float*)d_in[4];
    const float* norm1_b   = (const float*)d_in[5];
    const float* norm2_w   = (const float*)d_in[6];
    const float* norm2_b   = (const float*)d_in[7];
    const float* temperature = (const float*)d_in[8];
    const float* q_w       = (const float*)d_in[9];
    const float* q_dw      = (const float*)d_in[10];
    const float* kv_w      = (const float*)d_in[11];
    const float* kv_dw     = (const float*)d_in[12];
    const float* proj_w    = (const float*)d_in[13];
    const float* pin_w     = (const float*)d_in[14];
    const float* dw_w      = (const float*)d_in[15];
    const float* dw2_w     = (const float*)d_in[16];
    const float* pout_w    = (const float*)d_in[17];
    const float* ln1_w     = (const float*)d_in[18];
    const float* ln1_b     = (const float*)d_in[19];
    const float* ss_in_w   = (const float*)d_in[20];
    const float* ss_conv_w = (const float*)d_in[21];
    const float* ss_conv_b = (const float*)d_in[22];
    const float* ss_xproj_w= (const float*)d_in[23];
    const float* ss_dt_w   = (const float*)d_in[24];
    const float* ss_dt_b   = (const float*)d_in[25];
    const float* ss_Alogs  = (const float*)d_in[26];
    const float* ss_Ds     = (const float*)d_in[27];
    const float* ss_onorm_w= (const float*)d_in[28];
    const float* ss_onorm_b= (const float*)d_in[29];
    const float* ss_out_w  = (const float*)d_in[30];

    uint8_t* base = (uint8_t*)d_ws;
    size_t off = 0;
    auto alloc = [&](size_t bytes) -> void* {
        void* p = base + off;
        off = (off + bytes + 255) & ~(size_t)255;
        return p;
    };

    // ---- long-lived buffers ----
    float* Rcl    = (float*)alloc((size_t)PP * DIM * 4);
    float* r_cl   = (float*)alloc((size_t)PP * DIM * 4);
    float* dwout  = (float*)alloc((size_t)PP * DIN * 4);
    float* x2b    = (float*)alloc((size_t)PP * HID * 4);
    float* xxz    = (float*)alloc((size_t)PP * 2 * DIN * 4);
    float* xc     = (float*)alloc((size_t)PP * DIN * 4);
    float* x_dbl  = (float*)alloc((size_t)BB * KK * LL * 43 * 4);
    float* ycomb  = (float*)alloc((size_t)PP * DIN * 4);
    _Float16* ynz_h  = (_Float16*)alloc((size_t)PP * 352 * 2);
    float* ssout  = (float*)alloc((size_t)PP * HID * 4);
    _Float16* prod_h = (_Float16*)alloc((size_t)PP * 192 * 2);
    float* out_cl = (float*)alloc((size_t)PP * DIM * 4);

    // packed f16 weights (B operands padded to ceil64 rows)
    _Float16* w_conv1 = (_Float16*)alloc((size_t)64 * 32 * 2);
    _Float16* w_q     = (_Float16*)alloc((size_t)64 * 64 * 2);
    _Float16* w_kv    = (_Float16*)alloc((size_t)128 * 64 * 2);
    _Float16* w_proj  = (_Float16*)alloc((size_t)64 * 64 * 2);
    _Float16* w_pin   = (_Float16*)alloc((size_t)384 * 64 * 2);
    _Float16* w_in    = (_Float16*)alloc((size_t)704 * 192 * 2);
    _Float16* w_out   = (_Float16*)alloc((size_t)192 * 352 * 2);
    _Float16* w_pout  = (_Float16*)alloc((size_t)64 * 192 * 2);
    _Float16* w_xpw   = (_Float16*)alloc((size_t)BB * KK * 64 * 352 * 2);

    // ---- scratch region (reused across the three phases) ----
    const size_t scratch0 = off;

    // phase 1 (attention)
    _Float16* Xs_h   = (_Float16*)alloc((size_t)PP * 32 * 2);
    float*    s_cl   = (float*)alloc((size_t)PP * DIM * 4);
    _Float16* rn_h   = (_Float16*)alloc((size_t)PP * 64 * 2);
    _Float16* sn_h   = (_Float16*)alloc((size_t)PP * 64 * 2);
    float*    q0     = (float*)alloc((size_t)PP * 64 * 4);
    float*    kv0    = (float*)alloc((size_t)PP * 128 * 4);
    float*    q1     = (float*)alloc((size_t)PP * 64 * 4);
    float*    kv1    = (float*)alloc((size_t)PP * 128 * 4);
    // +32 pad rows so the 64-row B strip of the last batch stays in-bounds
    _Float16* qn_h   = (_Float16*)alloc((size_t)(BB * 64 + 32) * LL * 2);
    _Float16* kn_h   = (_Float16*)alloc((size_t)(BB * 64 + 32) * LL * 2);
    _Float16* vh     = (_Float16*)alloc((size_t)BB * 2 * LL * 32 * 2);
    float*    attnS  = (float*)alloc((size_t)BB * 2 * 32 * 32 * 4);
    _Float16* attnP  = (_Float16*)alloc((size_t)BB * 2 * 64 * 32 * 2);   // 64-row padded
    float*    g2out  = (float*)alloc((size_t)BB * 2 * LL * 32 * 4);
    _Float16* projin = (_Float16*)alloc((size_t)PP * 64 * 2);

    // phase 2 (ss2d prep) overlays phase 1
    off = scratch0;
    _Float16* t_h    = (_Float16*)alloc((size_t)PP * 64 * 2);
    float*    pinout = (float*)alloc((size_t)PP * DIN * 4);
    _Float16* rln_h  = (_Float16*)alloc((size_t)PP * 192 * 2);
    _Float16* xsh    = (_Float16*)alloc((size_t)BB * KK * LL * 352 * 2);

    // phase 3 (scan) overlays again
    off = scratch0;
    float* oy = (float*)alloc((size_t)BB * KK * LL * DIN * 4);
    (void)ws_size;

    const int TB = 256;
    auto g1 = [&](long long n) { return dim3((unsigned)cdiv(n, TB)); };

    // ===== weight packing =====
    k_nchw_to_nhwc<<<g1((long long)PP * DIM), TB, 0, stream>>>(input_R, Rcl);
    k_pack16<<<g1(64 * 32), TB, 0, stream>>>(conv1_w, 64, 32, 32, w_conv1, 64, 32);
    k_pack16<<<g1(64 * 64), TB, 0, stream>>>(q_w, 64, 64, 64, w_q, 64, 64);
    k_pack16<<<g1(128 * 64), TB, 0, stream>>>(kv_w, 128, 64, 64, w_kv, 128, 64);
    k_pack16<<<g1(64 * 64), TB, 0, stream>>>(proj_w, 64, 64, 64, w_proj, 64, 64);
    k_pack16<<<g1(384 * 64), TB, 0, stream>>>(pin_w, 2 * HID, 64, 64, w_pin, 384, 64);
    k_pack16<<<g1(704 * 192), TB, 0, stream>>>(ss_in_w, 2 * DIN, HID, HID, w_in, 704, 192);
    k_pack16<<<g1(192 * 352), TB, 0, stream>>>(ss_out_w, HID, DIN, DIN, w_out, 192, 352);
    k_pack16<<<g1(64 * 192), TB, 0, stream>>>(pout_w, 64, HID, HID, w_pout, 64, 192);
    k_pack_xpw<<<g1((long long)BB * KK * 64 * 352), TB, 0, stream>>>(ss_xproj_w, w_xpw);

    // ===== phase 1: guided-fusion channel attention =====
    k_up_pack<<<g1((long long)PP * 32), TB, 0, stream>>>(input_S, Xs_h);
    gemm(stream, Xs_h, w_conv1, s_cl, conv1_b, nullptr,
         PP, 64, 32, 32, 32, 64, 0, 0, 0, 0, 1);                       // conv1
    k_ln_pack<<<g1(PP), TB, 0, stream>>>(Rcl, 64, 64, norm1_w, norm1_b, 1e-6f, rn_h, 64, PP);
    k_ln_pack<<<g1(PP), TB, 0, stream>>>(s_cl, 64, 64, norm1_w, norm1_b, 1e-6f, sn_h, 64, PP);
    gemm(stream, rn_h, w_q, q0, nullptr, nullptr, PP, 64, 64, 64, 64, 64, 0, 0, 0, 0, 1);
    gemm(stream, sn_h, w_kv, kv0, nullptr, nullptr, PP, 128, 64, 64, 64, 128, 0, 0, 0, 0, 1);
    k_dw3x3<<<g1((long long)PP * 64), TB, 0, stream>>>(q0, 64, q_dw, nullptr, nullptr, 0, q1, 64, 64, 0);
    k_dw3x3<<<g1((long long)PP * 128), TB, 0, stream>>>(kv0, 128, kv_dw, nullptr, nullptr, 0, kv1, 128, 128, 0);
    k_l2n_pack<<<dim3(BB * 64), dim3(256), 0, stream>>>(q1, 64, 0, qn_h);
    k_l2n_pack<<<dim3(BB * 64), dim3(256), 0, stream>>>(kv1, 128, 0, kn_h);
    k_pack_v<<<g1((long long)BB * 2 * LL * 32), TB, 0, stream>>>(kv1, vh);
    gemm(stream, qn_h, kn_h, attnS, nullptr, nullptr,
         32, 32, LL, LL, LL, 32, 32LL * LL, 32LL * LL, 1024, 0, BB * 2); // q @ k^T
    k_softmax32<<<dim3(32), dim3(256), 0, stream>>>(attnS, temperature, attnP);
    gemm(stream, vh, attnP, g2out, nullptr, nullptr,
         LL, 32, 32, 32, 32, 32, (long long)LL * 32, 2048, (long long)LL * 32, 0, BB * 2);
    k_repack_heads<<<g1((long long)PP * 64), TB, 0, stream>>>(g2out, projin);
    gemm(stream, projin, w_proj, r_cl, nullptr, Rcl,
         PP, 64, 64, 64, 64, 64, 0, 0, 0, 0, 1);                       // r = R + proj

    // ===== phase 2: IEL / SS2D prep =====
    k_ln_pack<<<g1(PP), TB, 0, stream>>>(r_cl, 64, 64, norm2_w, norm2_b, 1e-6f, t_h, 64, PP);
    gemm(stream, t_h, w_pin, pinout, nullptr, nullptr,
         PP, 2 * HID, 64, 64, 64, 2 * HID, 0, 0, 0, 0, 1);             // pin 1x1
    k_dw3x3<<<g1((long long)PP * DIN), TB, 0, stream>>>(pinout, DIN, dw_w, nullptr, nullptr, 0,
                                                        dwout, DIN, DIN, 0);
    k_dw3x3<<<g1((long long)PP * HID), TB, 0, stream>>>(dwout + HID, DIN, dw2_w, nullptr,
                                                        dwout + HID, DIN, x2b, HID, HID, 2);
    k_ln_pack<<<g1(PP), TB, 0, stream>>>(dwout, DIN, HID, ln1_w, ln1_b, 1e-6f, rln_h, 192, PP);
    gemm(stream, rln_h, w_in, xxz, nullptr, nullptr,
         PP, 2 * DIN, 192, 192, 192, 2 * DIN, 0, 0, 0, 0, 1);          // in_proj
    k_dw3x3<<<g1((long long)PP * DIN), TB, 0, stream>>>(xxz, 2 * DIN, ss_conv_w, ss_conv_b,
                                                        nullptr, 0, xc, DIN, DIN, 1);
    k_pack_xs<<<g1((long long)BB * KK * LL * 352), TB, 0, stream>>>(xc, xsh);
    gemm(stream, xsh, w_xpw, x_dbl, nullptr, nullptr,
         LL, 43, 352, 352, 352, 43,
         (long long)LL * 352, 64LL * 352, (long long)LL * 43, 0, BB * KK); // x_dbl

    // ===== phase 3: selective scan =====
    k_selscan<<<dim3(BB * KK), dim3(384), 0, stream>>>(xc, x_dbl, ss_dt_w, ss_dt_b,
                                                       ss_Alogs, ss_Ds, oy);
    k_combine<<<g1((long long)PP * DIN), TB, 0, stream>>>(oy, ycomb);
    k_ln_silu_pack<<<g1(PP), TB, 0, stream>>>(ycomb, xxz, ss_onorm_w, ss_onorm_b, ynz_h);
    gemm(stream, ynz_h, w_out, ssout, nullptr, nullptr,
         PP, HID, 352, 352, 352, HID, 0, 0, 0, 0, 1);                  // out_proj
    k_eltfinal<<<g1((long long)PP * 192), TB, 0, stream>>>(ssout, dwout, x2b, prod_h);
    gemm(stream, prod_h, w_pout, out_cl, nullptr, r_cl,
         PP, 64, 192, 192, 192, 64, 0, 0, 0, 0, 1);                    // pout + residual
    k_nhwc_to_nchw<<<g1((long long)PP * DIM), TB, 0, stream>>>(out_cl, (float*)d_out);
}
